// LearnedLitho3D2_22282290332366
// MI455X (gfx1250) — matrix operand order repaired and checked by reading, not compiled, not run on hardware
//
#include <hip/hip_runtime.h>
#include <hip/hip_bf16.h>
#include <math.h>

// ---------------------------------------------------------------------------
// MI455X / gfx1250 fused litho pipeline.
// Heavy matmul work (pointwise 32x32 MLP layer, areawise 16->16 3x3 conv)
// runs on v_wmma_f32_16x16x32_f16.  Everything is staged through 3 ping-pong
// f32 buffers in d_ws (48MB) that stay resident in the 192MB L2.
// GELU is a branch-free erf approximation on v_rcp_f32/v_exp_f32 so the
// dominant per-pixel activation work is straight-line VALU/TRANS that can
// co-execute with the WMMA pipe (no EXEC-divergent libm erff).
// ---------------------------------------------------------------------------

typedef __attribute__((ext_vector_type(16))) _Float16 v16h;
typedef __attribute__((ext_vector_type(8)))  _Float16 v8h;
typedef __attribute__((ext_vector_type(8)))  float    v8f;

union AFrag { v16h v; v8h h[2]; };

#define IMG_H 1024
#define IMG_W 1024
#define NB    4
#define NPIX  (NB * IMG_H * IMG_W)

// workspace layout (bytes)
#define WS_ILLUM 0        // 13 f32 taps
#define WS_DIFF  512      // 101 f32 taps
#define WS_TFRAG 1024     // 1024 halfs : thresh W2 B-fragments (2 N-tiles)
#define WS_SFRAG 3072     // 1024 halfs : shrink W2 B-fragments
#define WS_AFRAG 5120     // 2560 halfs : area  W2 B-fragments (5 K-chunks)
#define WS_BUF   16384    // 3 x NPIX f32 ping-pong buffers

// Exact-GELU via branch-free erf (Abramowitz-Stegun 7.1.26, |err|<1.5e-7).
// ~13 straight-line VALU ops incl. v_rcp_f32 + v_exp_f32; no EXEC branches.
__device__ __forceinline__ float geluf(float x) {
    float z  = x * 0.70710678118654752f;
    float az = fabsf(z);
    float t  = __builtin_amdgcn_rcpf(fmaf(0.3275911f, az, 1.0f));
    float p  = fmaf(1.061405429f, t, -1.453152027f);
    p = fmaf(p, t, 1.421413741f);
    p = fmaf(p, t, -0.284496736f);
    p = fmaf(p, t, 0.254829592f);
    p = p * t;
    float e    = __expf(-az * az);
    float erfa = fmaf(-p, e, 1.0f);       // erf(|z|)
    float erfz = copysignf(erfa, z);
    return 0.5f * x * (1.0f + erfz);
}

// ---------------------------------------------------------------------------
// Setup: build gaussian taps + swizzle weight matrices into WMMA B-fragment
// lane order:  element(lane,h) = B[k][n],  n = lane&15,  k = (h>=8?16:0)+(lane>=16?8:0)+(h&7)
// ---------------------------------------------------------------------------
__global__ void setup_kernel(const float* __restrict__ sigp,
                             const float* __restrict__ tw2,
                             const float* __restrict__ sw2,
                             const float* __restrict__ aw2,
                             float* __restrict__ illum, float* __restrict__ diff,
                             _Float16* __restrict__ tfrag,
                             _Float16* __restrict__ sfrag,
                             _Float16* __restrict__ afrag) {
    int tid = threadIdx.x;
    if (tid == 0) {  // 13-tap, sigma = 0.2
        float s = 0.f, g[13];
        for (int i = 0; i < 13; ++i) {
            float x = (float)(i - 6) * 0.1f;
            g[i] = expf(-x * x / (2.0f * 0.2f * 0.2f));
            s += g[i];
        }
        for (int i = 0; i < 13; ++i) illum[i] = g[i] / s;
    }
    if (tid == 1) {  // 101-tap, sigma = sigmoid(p)*2.5
        float sg = 2.5f / (1.0f + expf(-sigp[0]));
        float s = 0.f, g[101];
        for (int i = 0; i < 101; ++i) {
            float x = (float)(i - 50) * 0.1f;
            g[i] = expf(-x * x / (2.0f * sg * sg));
            s += g[i];
        }
        for (int i = 0; i < 101; ++i) diff[i] = g[i] / s;
    }
    // W2 (32x32) -> two 16-col N-tiles of B fragments (B[k][n] = W2[n][k])
    for (int e = tid; e < 1024; e += blockDim.x) {
        int tile = e >> 9, r = e & 511, ln = r >> 4, h = r & 15;
        int n = ln & 15, k = ((h >> 3) << 4) + ((ln >> 4) << 3) + (h & 7);
        tfrag[e] = (_Float16)tw2[(tile * 16 + n) * 32 + k];
        sfrag[e] = (_Float16)sw2[(tile * 16 + n) * 32 + k];
    }
    // area W2 (16,16,3,3) -> 5 K-chunks of (2 taps x 16ch); chunk 4 pads tap 9
    for (int e = tid; e < 2560; e += blockDim.x) {
        int q = e / 512, r = e % 512, ln = r >> 4, h = r & 15;
        int n = ln & 15, kl = ((h >> 3) << 4) + ((ln >> 4) << 3) + (h & 7);
        int t = 2 * q + (kl >= 16 ? 1 : 0), ch = kl & 15;
        afrag[e] = (t <= 8) ? (_Float16)aw2[(n * 16 + ch) * 9 + t] : (_Float16)0.f;
    }
}

// ---------------------------------------------------------------------------
// Separable gaussian, horizontal pass (zero-padded). scale folds x/100.
// ---------------------------------------------------------------------------
__global__ void blur_row_kernel(const float* __restrict__ src, float* __restrict__ dst,
                                const float* __restrict__ taps, int R, float scale) {
    __shared__ float seg[256 + 100];
    int tid = threadIdx.x;
    int x0 = blockIdx.x * 256;
    long base = ((long)blockIdx.z * IMG_H + blockIdx.y) * IMG_W;
    __builtin_prefetch(src + base + x0 + tid, 0, 1);   // global_prefetch_b8
    int cnt = 256 + 2 * R;
    for (int i = tid; i < cnt; i += 256) {
        int gx = x0 - R + i;
        seg[i] = (gx >= 0 && gx < IMG_W) ? src[base + gx] * scale : 0.f;
    }
    __syncthreads();
    float acc = 0.f;
    for (int t = 0; t <= 2 * R; ++t) acc = fmaf(taps[t], seg[tid + t], acc);
    dst[base + x0 + tid] = acc;
}

// Vertical pass: 32 cols x 64 rows per block, coalesced loads into LDS.
__global__ void blur_col_kernel(const float* __restrict__ src, float* __restrict__ dst,
                                const float* __restrict__ taps, int R) {
    __shared__ float tile[164][32];
    int tx = threadIdx.x, ty = threadIdx.y;
    int tid = ty * 32 + tx;
    int x0 = blockIdx.x * 32, y0 = blockIdx.y * 64;
    long ib = (long)blockIdx.z * IMG_H * IMG_W;
    __builtin_prefetch(src + ib + (long)y0 * IMG_W + x0 + tx, 0, 1);
    int rows = 64 + 2 * R;
    for (int e = tid; e < rows * 32; e += 256) {
        int ry = e >> 5, rx = e & 31;
        int gy = y0 - R + ry;
        tile[ry][rx] = (gy >= 0 && gy < IMG_H) ? src[ib + (long)gy * IMG_W + x0 + rx] : 0.f;
    }
    __syncthreads();
    for (int i = 0; i < 8; ++i) {
        int oy = ty + i * 8;
        float acc = 0.f;
        for (int t = 0; t <= 2 * R; ++t) acc = fmaf(taps[t], tile[oy + t][tx], acc);
        dst[ib + (long)(y0 + oy) * IMG_W + x0 + tx] = acc;
    }
}

// ---------------------------------------------------------------------------
// Pointwise MLP 1->32->32->1 (exact GELU).  Layer 2 = v_wmma_f32_16x16x32_f16:
// A = 16 pixels x 32ch of gelu(w1*s+b1) built per-lane to the ISA A layout,
// B = pre-swizzled W2 fragments, C f32.  Layer 3 via half-wave shfl reduce.
// Each wave owns 8 whole M-tiles (EXEC stays all-ones for WMMA).
// ---------------------------------------------------------------------------
__global__ void pointwise_kernel(const float* __restrict__ src, float* __restrict__ dst,
                                 const float* __restrict__ w1, const float* __restrict__ b1,
                                 const _Float16* __restrict__ frag,
                                 const float* __restrict__ b2, const float* __restrict__ w3,
                                 const float* __restrict__ b3,
                                 const float* __restrict__ mul) {
    __shared__ float w1s[32], b1s[32], b2s[32], w3s[32];
    int tid = threadIdx.x;
    if (tid < 32) { w1s[tid] = w1[tid]; b1s[tid] = b1[tid]; b2s[tid] = b2[tid]; w3s[tid] = w3[tid]; }
    __syncthreads();

    int lane = tid & 31, wave = tid >> 5;
    int hi = lane >> 4;
    int n = lane & 15;
    float b3v = b3[0];

    v16h bf0 = *(const v16h*)(frag + lane * 16);
    v16h bf1 = *(const v16h*)(frag + 512 + lane * 16);

    long waveTile0 = ((long)blockIdx.x * 8 + wave) * 8;
#pragma unroll 1
    for (int j = 0; j < 8; ++j) {
        long pix0 = (waveTile0 + j) << 4;
        float s = src[pix0 + (lane & 15)];  // m = lane&15 (both half-waves)

        v16h a = {};
#pragma unroll
        for (int h = 0; h < 16; ++h) {  // ISA A layout: k = h + (h>=8)*8 + hi*8
            int k = h + (h >= 8 ? 8 : 0) + hi * 8;
            a[h] = (_Float16)geluf(fmaf(w1s[k], s, b1s[k]));
        }
        v8f z = {};
        v8f c0 = __builtin_amdgcn_wmma_f32_16x16x32_f16(false, a, false, bf0, (short)0, z, false, false);
        v8f c1 = __builtin_amdgcn_wmma_f32_16x16x32_f16(false, a, false, bf1, (short)0, z, false, false);

        float y = 0.f;
#pragma unroll
        for (int r = 0; r < 8; ++r) {  // C row r -> pixel m = r + hi*8, this lane holds ch n
            float g0 = geluf(c0[r] + b2s[n]);
            float g1 = geluf(c1[r] + b2s[n + 16]);
            float cr = fmaf(w3s[n], g0, w3s[n + 16] * g1);
            cr += __shfl_xor(cr, 1, 16);
            cr += __shfl_xor(cr, 2, 16);
            cr += __shfl_xor(cr, 4, 16);
            cr += __shfl_xor(cr, 8, 16);
            if ((lane & 15) == r) y = cr;
        }
        if ((lane & 15) < 8) {
            int m = (lane & 15) + hi * 8;
            float o = y + b3v;
            if (mul) o *= mul[pix0 + m];
            dst[pix0 + m] = o;
        }
    }
}

// ---------------------------------------------------------------------------
// Areawise 3x3 CNN 1->16->16->1, tile 30x16 out (+3 halo each side).
// Middle conv = 5 WMMAs per 16-pixel row tile (K = 2 taps x 16 ch, chunk 4
// zero-padded), A gathered from LDS f16 h1 tile with 16B ds_load_b128s.
// ---------------------------------------------------------------------------
__global__ void areawise_kernel(const float* __restrict__ src, float* __restrict__ out,
                                const float* __restrict__ aw1, const float* __restrict__ ab1,
                                const _Float16* __restrict__ afrag,
                                const float* __restrict__ ab2, const float* __restrict__ aw3,
                                const float* __restrict__ ab3) {
    __shared__ float tIn[22][36];
    __shared__ __align__(16) _Float16 h1[20][34][16];
    __shared__ __align__(16) _Float16 h2g[18][32][16];
    __shared__ float w1s[144], b1s[16], b2s[16], w3s[144];

    int tid = threadIdx.x;
    int lane = tid & 31, wave = tid >> 5;
    int hi = lane >> 4, n = lane & 15;
    int ox0 = blockIdx.x * 30, oy0 = blockIdx.y * 16;
    long ib = (long)blockIdx.z * IMG_H * IMG_W;

    if (tid < 144) { w1s[tid] = aw1[tid]; w3s[tid] = aw3[tid]; }
    if (tid < 16)  { b1s[tid] = ab1[tid]; b2s[tid] = ab2[tid]; }

    v16h bq[5];
#pragma unroll
    for (int q = 0; q < 5; ++q) bq[q] = *(const v16h*)(afrag + q * 512 + lane * 16);

    // phase 1: load t tile (zero-padded)
    for (int e = tid; e < 22 * 36; e += 256) {
        int j = e / 36, i = e % 36;
        int gy = oy0 - 3 + j, gx = ox0 - 3 + i;
        tIn[j][i] = (gy >= 0 && gy < IMG_H && gx >= 0 && gx < IMG_W)
                        ? src[ib + (long)gy * IMG_W + gx] : 0.f;
    }
    __syncthreads();

    // phase 2: h1 = gelu(conv3(t, w1)+b1); force 0 outside image ('SAME' pad)
    for (int e = tid; e < 20 * 34; e += 256) {
        int j = e / 34, i = e % 34;
        int gy = oy0 - 2 + j, gx = ox0 - 2 + i;
        bool inimg = (gy >= 0 && gy < IMG_H && gx >= 0 && gx < IMG_W);
#pragma unroll
        for (int c = 0; c < 16; ++c) {
            float acc = b1s[c];
#pragma unroll
            for (int ky = 0; ky < 3; ++ky)
#pragma unroll
                for (int kx = 0; kx < 3; ++kx)
                    acc = fmaf(w1s[c * 9 + ky * 3 + kx], tIn[j + ky][i + kx], acc);
            h1[j][i][c] = (_Float16)(inimg ? geluf(acc) : 0.f);
        }
    }
    __syncthreads();

    // phase 3: h2 via WMMA. 36 M-tiles (18 rows x 2 halves), wave-uniform loop.
    int m = lane & 15, chb = hi * 8;
    for (int tt = wave; tt < 36; tt += 8) {
        int r = tt >> 1, xoff = (tt & 1) * 16;
        int xp = xoff + m;
        v8f c = {};
        AFrag a;
#pragma unroll
        for (int q = 0; q < 5; ++q) {
            int t0 = 2 * q, t1 = t0 + 1;
            a.h[0] = *(const v8h*)&h1[r + t0 / 3][xp + t0 % 3][chb];
            if (t1 <= 8) a.h[1] = *(const v8h*)&h1[r + t1 / 3][xp + t1 % 3][chb];
            else { v8h zz = {}; a.h[1] = zz; }
            c = __builtin_amdgcn_wmma_f32_16x16x32_f16(false, a.v, false, bq[q], (short)0, c, false, false);
        }
#pragma unroll
        for (int rr = 0; rr < 8; ++rr) {  // C: lane holds ch n, row rr -> pixel rr+hi*8
            int xi = xoff + rr + hi * 8;
            int gy = oy0 - 1 + r, gx = ox0 - 1 + xi;
            bool inimg = (gy >= 0 && gy < IMG_H && gx >= 0 && gx < IMG_W);
            h2g[r][xi][n] = (_Float16)(inimg ? geluf(c[rr] + b2s[n]) : 0.f);
        }
    }
    __syncthreads();

    // phase 4: out = conv3(h2g, w3) + b3, x100
    float b3v = ab3[0];
    for (int e = tid; e < 30 * 16; e += 256) {
        int jj = e / 30, ii = e % 30;
        float acc = b3v;
#pragma unroll
        for (int o = 0; o < 16; ++o)
#pragma unroll
            for (int ky = 0; ky < 3; ++ky)
#pragma unroll
                for (int kx = 0; kx < 3; ++kx)
                    acc = fmaf(w3s[o * 9 + ky * 3 + kx],
                               (float)h2g[jj + ky][ii + kx][o], acc);
        int gy = oy0 + jj, gx = ox0 + ii;
        if (gy < IMG_H && gx < IMG_W)
            out[ib + (long)gy * IMG_W + gx] = acc * 100.f;
    }
}

// ---------------------------------------------------------------------------
extern "C" void kernel_launch(void* const* d_in, const int* in_sizes, int n_in,
                              void* d_out, int out_size, void* d_ws, size_t ws_size,
                              hipStream_t stream) {
    const float* masks = (const float*)d_in[0];
    const float* sigp  = (const float*)d_in[1];
    // thresh_params: w1,b1,w2,b2,w3,b3
    const float* tw1 = (const float*)d_in[2];  const float* tb1 = (const float*)d_in[3];
    const float* tw2 = (const float*)d_in[4];  const float* tb2 = (const float*)d_in[5];
    const float* tw3 = (const float*)d_in[6];  const float* tb3 = (const float*)d_in[7];
    // shrink_params
    const float* sw1 = (const float*)d_in[8];  const float* sb1 = (const float*)d_in[9];
    const float* sw2 = (const float*)d_in[10]; const float* sb2 = (const float*)d_in[11];
    const float* sw3 = (const float*)d_in[12]; const float* sb3 = (const float*)d_in[13];
    // out_params
    const float* aw1 = (const float*)d_in[14]; const float* ab1 = (const float*)d_in[15];
    const float* aw2 = (const float*)d_in[16]; const float* ab2 = (const float*)d_in[17];
    const float* aw3 = (const float*)d_in[18]; const float* ab3 = (const float*)d_in[19];

    char* ws = (char*)d_ws;
    float*    illum = (float*)(ws + WS_ILLUM);
    float*    diff  = (float*)(ws + WS_DIFF);
    _Float16* tfrag = (_Float16*)(ws + WS_TFRAG);
    _Float16* sfrag = (_Float16*)(ws + WS_SFRAG);
    _Float16* afrag = (_Float16*)(ws + WS_AFRAG);
    float* bufA = (float*)(ws + WS_BUF);
    float* bufB = bufA + NPIX;
    float* bufC = bufB + NPIX;
    float* outp = (float*)d_out;

    setup_kernel<<<1, 128, 0, stream>>>(sigp, tw2, sw2, aw2, illum, diff, tfrag, sfrag, afrag);

    dim3 rowGrid(IMG_W / 256, IMG_H, NB);
    dim3 colGrid(IMG_W / 32, IMG_H / 64, NB);
    dim3 colBlk(32, 8);

    // x/100 -> 13-tap separable blur -> bufB
    blur_row_kernel<<<rowGrid, 256, 0, stream>>>(masks, bufA, illum, 6, 0.01f);
    blur_col_kernel<<<colGrid, colBlk, 0, stream>>>(bufA, bufB, illum, 6);
    // exposure = pointwise(bufB, thresh) -> bufC
    pointwise_kernel<<<NPIX / 1024, 256, 0, stream>>>(bufB, bufC, tw1, tb1, tfrag, tb2, tw3, tb3, nullptr);
    // diffusion = 101-tap separable blur(exposure) -> bufB
    blur_row_kernel<<<rowGrid, 256, 0, stream>>>(bufC, bufA, diff, 50, 1.0f);
    blur_col_kernel<<<colGrid, colBlk, 0, stream>>>(bufA, bufB, diff, 50);
    // t = exposure * pointwise(diffusion, shrink) -> bufA
    pointwise_kernel<<<NPIX / 1024, 256, 0, stream>>>(bufB, bufA, sw1, sb1, sfrag, sb2, sw3, sb3, bufC);
    // out = areawise(t) * 100
    dim3 areaGrid((IMG_W + 29) / 30, IMG_H / 16, NB);
    areawise_kernel<<<areaGrid, 256, 0, stream>>>(bufA, outp, aw1, ab1, afrag, ab2, aw3, ab3);
}
